// MultiGATv2_21182778703917
// MI455X (gfx1250) — matrix operand order repaired and checked by reading, not compiled, not run on hardware
//
#include <hip/hip_runtime.h>
#include <hip/hip_bf16.h>
#include <math.h>

#define DIN       128
#define NHEAD     3
#define FDIM      64
#define HF        192          // NHEAD * FDIM
#define NEG_SLOPE 0.2f

typedef _Float16 v8h  __attribute__((ext_vector_type(8)));
typedef _Float16 v16h __attribute__((ext_vector_type(16)));
typedef float    v8f  __attribute__((ext_vector_type(8)));

// ---- order-preserving float <-> u32 key for atomic segment-max -------------
__device__ __forceinline__ unsigned fkey(float f) {
  unsigned u = __float_as_uint(f);
  return (u & 0x80000000u) ? ~u : (u | 0x80000000u);
}
__device__ __forceinline__ float funkey(unsigned k) {
  unsigned u = (k & 0x80000000u) ? (k ^ 0x80000000u) : ~k;
  return __uint_as_float(u);
}

// ---- elementwise f32 -> f16 ------------------------------------------------
__global__ void k_f32_to_f16(const float* __restrict__ x, _Float16* __restrict__ y,
                             long n) {
  long i = (long)blockIdx.x * blockDim.x + threadIdx.x;
  if (i < n) y[i] = (_Float16)x[i];
}

// ---- pre-swizzle W (f32 [K,192] row-major) into WMMA B-fragment order ------
// layout: Bfrag[mat(2)][ct(12)][kc(K/32)][lane(32)][half(16)]
// lane l: col n = ct*16 + (l&15); half j -> K = kc*32 + ((l>>4)*8) + (j&7) + ((j>>3)*16)
__global__ void k_prep_w(const float* __restrict__ Ws, const float* __restrict__ Wd,
                         _Float16* __restrict__ Bfrag, int K) {
  int KC = K >> 5;
  int total = 2 * 12 * KC * 32 * 16;
  int i = blockIdx.x * blockDim.x + threadIdx.x;
  if (i >= total) return;
  int j    = i & 15;
  int lane = (i >> 4) & 31;
  int t    = i >> 9;                 // [mat][ct][kc]
  int kc   = t % KC;
  int ct   = (t / KC) % 12;
  int mat  = t / (KC * 12);
  int col  = ct * 16 + (lane & 15);
  int k    = kc * 32 + ((lane >> 4) << 3) + (j & 7) + ((j >> 3) << 4);
  const float* W = mat ? Wd : Ws;
  Bfrag[i] = (_Float16)W[(size_t)k * HF + col];
}

// ---- zero accumulators -----------------------------------------------------
__global__ void k_init(float* __restrict__ U, float* __restrict__ denom,
                       unsigned* __restrict__ mkey, int Nn) {
  long i = (long)blockIdx.x * blockDim.x + threadIdx.x;
  if (i < (long)Nn * HF) U[i] = 0.0f;
  if (i < (long)Nn * NHEAD) { denom[i] = 0.0f; mkey[i] = 0u; }
}

// ---- fragment load / tile store helpers ------------------------------------
__device__ __forceinline__ v16h ldfrag(const _Float16* p) {   // B: 32B contig
  v8h lo = *(const v8h*)p;
  v8h hi = *(const v8h*)(p + 8);
  return __builtin_shufflevector(lo, hi, 0,1,2,3,4,5,6,7,8,9,10,11,12,13,14,15);
}
__device__ __forceinline__ v16h ldfragA(const _Float16* p) {  // A: 2x16B @ +0,+32B
  v8h lo = *(const v8h*)p;
  v8h hi = *(const v8h*)(p + 16);
  return __builtin_shufflevector(lo, hi, 0,1,2,3,4,5,6,7,8,9,10,11,12,13,14,15);
}

__device__ __forceinline__ void store_tile(float* __restrict__ out,
                                           const float* __restrict__ bias,
                                           int ct, int ncol, int mbase, int Nrows,
                                           v8f c) {
  int n = ct * 16 + ncol;
  float b = bias[n];
  #pragma unroll
  for (int r = 0; r < 8; ++r) {
    int m = mbase + r;
    if (m < Nrows) out[(size_t)m * HF + n] = c[r] + b;
  }
}

// ---- WMMA GEMM: fs = Xh@Ws + bs, fd = Xh@Wd + bd ---------------------------
// One wave computes 4 output tiles (2 matrices x 2 adjacent column tiles)
// sharing one A fragment per k-chunk. Software double-buffered: the k+1 chunk
// loads are issued before the k chunk WMMAs so waits are partial.
__global__ void k_gemm_wmma(const _Float16* __restrict__ Xh,      // [N,K] f16
                            const _Float16* __restrict__ Bfrag,   // swizzled W
                            const float* __restrict__ bs, const float* __restrict__ bd,
                            float* __restrict__ fs, float* __restrict__ fd,
                            int Nrows, int K) {
  const int KC = K >> 5;
  int gtid = blockIdx.x * blockDim.x + threadIdx.x;
  int wave = gtid >> 5;
  int lane = gtid & 31;
  int rtiles = (Nrows + 15) >> 4;
  if (wave >= rtiles * 6) return;
  int rt  = wave / 6;
  int cg  = wave - rt * 6;         // column group -> cts {2cg, 2cg+1}
  int ct0 = cg * 2, ct1 = ct0 + 1;

  // A fragment (ISA 16-bit A 16x32 layout): lane row = l&15,
  // halfs 0..7 = K kb..kb+7, halfs 8..15 = K 16+kb..16+kb+7, kb = (l>>4)*8
  int row = rt * 16 + (lane & 15);
  if (row >= Nrows) row = Nrows - 1;          // clamped load; store is guarded
  int kb = (lane >> 4) << 3;
  const _Float16* xr = Xh + (size_t)row * K + kb;

  size_t fragsz = (size_t)KC * 32 * 16;       // halfs per (mat,ct) fragment set
  const _Float16* bS0 = Bfrag + (size_t)(0 * 12 + ct0) * fragsz + lane * 16;
  const _Float16* bS1 = Bfrag + (size_t)(0 * 12 + ct1) * fragsz + lane * 16;
  const _Float16* bD0 = Bfrag + (size_t)(1 * 12 + ct0) * fragsz + lane * 16;
  const _Float16* bD1 = Bfrag + (size_t)(1 * 12 + ct1) * fragsz + lane * 16;

  v8f cs0 = {}, cs1 = {}, cd0 = {}, cd1 = {};

  // prologue: load chunk 0
  v16h a  = ldfragA(xr);
  v16h b0 = ldfrag(bS0);
  v16h b1 = ldfrag(bS1);
  v16h b2 = ldfrag(bD0);
  v16h b3 = ldfrag(bD1);

  for (int kc = 0; kc < KC; ++kc) {
    // issue next-chunk loads first (clamped on last iteration; values unused)
    int adv = (kc + 1 < KC) ? 1 : 0;
    const _Float16* xrn  = xr  + adv * 32;
    const _Float16* bS0n = bS0 + adv * 512;
    const _Float16* bS1n = bS1 + adv * 512;
    const _Float16* bD0n = bD0 + adv * 512;
    const _Float16* bD1n = bD1 + adv * 512;
    v16h an  = ldfragA(xrn);
    v16h bn0 = ldfrag(bS0n);
    v16h bn1 = ldfrag(bS1n);
    v16h bn2 = ldfrag(bD0n);
    v16h bn3 = ldfrag(bD1n);

    // compute on current chunk while next-chunk loads are in flight
    cs0 = __builtin_amdgcn_wmma_f32_16x16x32_f16(false, a, false, b0,
                                                 (short)0, cs0, false, false);
    cs1 = __builtin_amdgcn_wmma_f32_16x16x32_f16(false, a, false, b1,
                                                 (short)0, cs1, false, false);
    cd0 = __builtin_amdgcn_wmma_f32_16x16x32_f16(false, a, false, b2,
                                                 (short)0, cd0, false, false);
    cd1 = __builtin_amdgcn_wmma_f32_16x16x32_f16(false, a, false, b3,
                                                 (short)0, cd1, false, false);

    a = an; b0 = bn0; b1 = bn1; b2 = bn2; b3 = bn3;
    xr = xrn; bS0 = bS0n; bS1 = bS1n; bD0 = bD0n; bD1 = bD1n;
  }

  // D layout: lane col n = l&15; VGPR r -> row r + 8*(l>=16)
  int ncol  = lane & 15;
  int mbase = rt * 16 + ((lane >> 4) << 3);
  store_tile(fs, bs, ct0, ncol, mbase, Nrows, cs0);
  store_tile(fs, bs, ct1, ncol, mbase, Nrows, cs1);
  store_tile(fd, bd, ct0, ncol, mbase, Nrows, cd0);
  store_tile(fd, bd, ct1, ncol, mbase, Nrows, cd1);
}

// ---- edge pass 1: logits[e][h] = attn . leakyrelu(fs[src]+fd[dst]); seg-max -
__global__ void k_edge_logits_max(const float* __restrict__ fs,
                                  const float* __restrict__ fd,
                                  const int* __restrict__ src, const int* __restrict__ dst,
                                  const float* __restrict__ attn,   // [192]
                                  float* __restrict__ logits,       // [E,3]
                                  unsigned* __restrict__ mkey, int E) {
  int gtid = blockIdx.x * blockDim.x + threadIdx.x;
  int e = gtid >> 5, lane = gtid & 31;
  if (e >= E) return;
  int s = src[e], d = dst[e];
  // prefetch a future edge's rows to hide the dependent-gather latency
  int ep = e + 256;
  if (ep < E) {
    int sp = src[ep], dp = dst[ep];
    __builtin_prefetch(fs + (size_t)sp * HF + lane * 4, 0, 0);
    __builtin_prefetch(fd + (size_t)dp * HF + lane * 4, 0, 0);
  }
  const float* ps = fs + (size_t)s * HF;
  const float* pd = fd + (size_t)d * HF;
  float p0 = 0.f, p1 = 0.f, p2 = 0.f;
  #pragma unroll
  for (int t = 0; t < 6; ++t) {
    int cidx = lane + 32 * t;
    float v = ps[cidx] + pd[cidx];
    v = v > 0.f ? v : NEG_SLOPE * v;
    float term = attn[cidx] * v;
    if (t < 2) p0 += term; else if (t < 4) p1 += term; else p2 += term;
  }
  #pragma unroll
  for (int off = 16; off; off >>= 1) {
    p0 += __shfl_xor(p0, off, 32);
    p1 += __shfl_xor(p1, off, 32);
    p2 += __shfl_xor(p2, off, 32);
  }
  if (lane == 0) {
    logits[(size_t)e * 3 + 0] = p0;
    logits[(size_t)e * 3 + 1] = p1;
    logits[(size_t)e * 3 + 2] = p2;
    atomicMax(&mkey[d * 3 + 0], fkey(p0));
    atomicMax(&mkey[d * 3 + 1], fkey(p1));
    atomicMax(&mkey[d * 3 + 2], fkey(p2));
  }
}

// ---- edge pass 2: ex = exp(logit - max); denom += ex; U[dst] += ex*fs[src] --
__global__ void k_edge_accum(const float* __restrict__ fs,
                             const int* __restrict__ src, const int* __restrict__ dst,
                             const float* __restrict__ logits,
                             const unsigned* __restrict__ mkey,
                             float* __restrict__ denom, float* __restrict__ U, int E) {
  int gtid = blockIdx.x * blockDim.x + threadIdx.x;
  int e = gtid >> 5, lane = gtid & 31;
  if (e >= E) return;
  int s = src[e], d = dst[e];
  int ep = e + 256;
  if (ep < E) {
    int sp = src[ep];
    __builtin_prefetch(fs + (size_t)sp * HF + lane * 4, 0, 0);
  }
  float ex0 = expf(logits[(size_t)e * 3 + 0] - funkey(mkey[d * 3 + 0]));
  float ex1 = expf(logits[(size_t)e * 3 + 1] - funkey(mkey[d * 3 + 1]));
  float ex2 = expf(logits[(size_t)e * 3 + 2] - funkey(mkey[d * 3 + 2]));
  if (lane == 0) {
    atomicAdd(&denom[d * 3 + 0], ex0);
    atomicAdd(&denom[d * 3 + 1], ex1);
    atomicAdd(&denom[d * 3 + 2], ex2);
  }
  const float* ps = fs + (size_t)s * HF;
  float* pu = U + (size_t)d * HF;
  #pragma unroll
  for (int t = 0; t < 6; ++t) {
    int cidx = lane + 32 * t;
    float ex = (t < 2) ? ex0 : (t < 4) ? ex1 : ex2;
    atomicAdd(&pu[cidx], ex * ps[cidx]);
  }
}

// ---- node finalize: out = tanh(U/denom + bias), in place -------------------
__global__ void k_finalize(float* __restrict__ U, const float* __restrict__ denom,
                           const float* __restrict__ bias, int Nn) {
  long i = (long)blockIdx.x * blockDim.x + threadIdx.x;
  if (i >= (long)Nn * HF) return;
  int node = (int)(i / HF);
  int cidx = (int)(i - (long)node * HF);
  int h = cidx >> 6;
  float dn = denom[node * 3 + h];
  float v = (dn > 0.f) ? (U[i] / dn) : 0.f;
  U[i] = tanhf(v + bias[cidx]);
}

// ---------------------------------------------------------------------------
static void run_layer(const float* X, int K,
                      const float* Ws, const float* bs,
                      const float* Wd, const float* bd,
                      const float* attn, const float* bias,
                      const int* src, const int* dst, int N, int E,
                      float* out,                       // [N,HF] dest & U accum
                      _Float16* Xh, _Float16* Bfrag, float* fs, float* fd,
                      float* logits, unsigned* mkey, float* denom,
                      hipStream_t stream) {
  long nX = (long)N * K;
  k_f32_to_f16<<<(int)((nX + 255) / 256), 256, 0, stream>>>(X, Xh, nX);
  int KC = K >> 5;
  int prepTot = 2 * 12 * KC * 32 * 16;
  k_prep_w<<<(prepTot + 255) / 256, 256, 0, stream>>>(Ws, Wd, Bfrag, K);
  long nU = (long)N * HF;
  k_init<<<(int)((nU + 255) / 256), 256, 0, stream>>>(out, denom, mkey, N);
  int rtiles = (N + 15) >> 4;
  long gthreads = (long)rtiles * 6 * 32;
  k_gemm_wmma<<<(int)((gthreads + 255) / 256), 256, 0, stream>>>(
      Xh, Bfrag, bs, bd, fs, fd, N, K);
  long ethreads = (long)E * 32;
  int eblocks = (int)((ethreads + 255) / 256);
  k_edge_logits_max<<<eblocks, 256, 0, stream>>>(fs, fd, src, dst, attn,
                                                 logits, mkey, E);
  k_edge_accum<<<eblocks, 256, 0, stream>>>(fs, src, dst, logits, mkey,
                                            denom, out, E);
  k_finalize<<<(int)((nU + 255) / 256), 256, 0, stream>>>(out, denom, bias, N);
}

extern "C" void kernel_launch(void* const* d_in, const int* in_sizes, int n_in,
                              void* d_out, int out_size, void* d_ws, size_t ws_size,
                              hipStream_t stream) {
  const float* feats = (const float*)d_in[0];
  const int*   src   = (const int*)d_in[1];
  const int*   dst   = (const int*)d_in[2];
  const float* W1s = (const float*)d_in[3];
  const float* b1s = (const float*)d_in[4];
  const float* W1d = (const float*)d_in[5];
  const float* b1d = (const float*)d_in[6];
  const float* attn1 = (const float*)d_in[7];
  const float* bias1 = (const float*)d_in[8];
  const float* W2s = (const float*)d_in[9];
  const float* b2s = (const float*)d_in[10];
  const float* W2d = (const float*)d_in[11];
  const float* b2d = (const float*)d_in[12];
  const float* attn2 = (const float*)d_in[13];
  const float* bias2 = (const float*)d_in[14];

  int N = in_sizes[0] / DIN;
  int E = in_sizes[1];

  float* out0 = (float*)d_out;
  float* out1 = out0 + (size_t)N * DIN;
  float* out2 = out1 + (size_t)N * HF;

  // carve workspace (bump allocator, 256B aligned)
  char* w = (char*)d_ws;
  auto carve = [&](size_t bytes) -> void* {
    void* p = (void*)w;
    w += (bytes + 255) & ~(size_t)255;
    return p;
  };
  _Float16* Xh     = (_Float16*)carve((size_t)N * HF * sizeof(_Float16));
  _Float16* Bfrag  = (_Float16*)carve((size_t)2 * 12 * 6 * 32 * 16 * sizeof(_Float16));
  float*    fs     = (float*)carve((size_t)N * HF * sizeof(float));
  float*    fd     = (float*)carve((size_t)N * HF * sizeof(float));
  float*    logits = (float*)carve((size_t)E * NHEAD * sizeof(float));
  unsigned* mkey   = (unsigned*)carve((size_t)N * NHEAD * sizeof(unsigned));
  float*    denom  = (float*)carve((size_t)N * NHEAD * sizeof(float));

  // out0 = feats
  hipMemcpyAsync(out0, feats, sizeof(float) * (size_t)N * DIN,
                 hipMemcpyDeviceToDevice, stream);

  // layer 1: K = 128, input feats -> out1
  run_layer(feats, DIN, W1s, b1s, W1d, b1d, attn1, bias1, src, dst, N, E,
            out1, Xh, Bfrag, fs, fd, logits, mkey, denom, stream);
  // layer 2: K = 192, input out1 -> out2
  run_layer(out1, HF, W2s, b2s, W2d, b2d, attn2, bias2, src, dst, N, E,
            out2, Xh, Bfrag, fs, fd, logits, mkey, denom, stream);
}
